// Decoder_50259707298692
// MI455X (gfx1250) — compile-verified, matrix operand order checked by reading, not run on hardware
//
#include <hip/hip_runtime.h>
#include <math.h>

// ---------------- problem constants ----------------
constexpr int Bb   = 128;
constexpr int Tt   = 512;
constexpr int Hh   = 512;
constexpr int Ee   = 512;
constexpr int Vout = 30000;

typedef __attribute__((ext_vector_type(16))) __bf16 v16bf;
typedef __attribute__((ext_vector_type(8)))  __bf16 v8bf;
typedef __attribute__((ext_vector_type(8)))  float  v8f;

// ---------------- workspace layout (bytes) ----------------
constexpr size_t WS_W2C  = 0;                         // [H*H] bf16  (code enc-proj weight)
constexpr size_t WS_W2A  = WS_W2C  + Hh*Hh*2;         // [H*H] bf16
constexpr size_t WS_HPC  = WS_W2A  + Hh*Hh*2;         // [B*H] f32   (h_proj + bias, code)
constexpr size_t WS_HPA  = WS_HPC  + Bb*Hh*4;         // [B*H] f32
constexpr size_t WS_W3C  = WS_HPA  + Bb*Hh*4;         // [H]   f32   (W[:,2H:] @ Wc, code)
constexpr size_t WS_W3A  = WS_W3C  + Hh*4;            // [H]   f32
constexpr size_t WS_SCC  = WS_W3A  + Hh*4;            // [B*T] f32   scores code
constexpr size_t WS_SCA  = WS_SCC  + Bb*Tt*4;         // [B*T] f32   scores ast
constexpr size_t WS_CTXC = WS_SCA  + Bb*Tt*4;         // [B*H] f32
constexpr size_t WS_CTXA = WS_CTXC + Bb*Hh*4;         // [B*H] f32
constexpr size_t WS_CTX  = WS_CTXA + Bb*Hh*4;         // [B*H] f32   combined ctx
constexpr size_t WS_XBF  = WS_CTX  + Bb*Hh*4;         // [B*2H] bf16 concat(h1,ctx)

// ---------------- helpers ----------------
// A/B fragment for v_wmma_f32_16x16x32_bf16 (ISA 7.12.2, 16-bit A 16x32):
// lane m = lane&15, half = lane>>4; elems 0..7 -> K = 8*half + 0..7,
// elems 8..15 -> K = 8*half + 16..23.  Both runs are contiguous in memory
// for a row-major [*, K] source, so load two 8-float runs and convert.
__device__ __forceinline__ v16bf pack_f32_frag(const float* __restrict__ p) {
  float4 f0 = *(const float4*)(p + 0);
  float4 f1 = *(const float4*)(p + 4);
  float4 f2 = *(const float4*)(p + 16);
  float4 f3 = *(const float4*)(p + 20);
  v16bf r;
  r[0]=(__bf16)f0.x; r[1]=(__bf16)f0.y; r[2]=(__bf16)f0.z;  r[3]=(__bf16)f0.w;
  r[4]=(__bf16)f1.x; r[5]=(__bf16)f1.y; r[6]=(__bf16)f1.z;  r[7]=(__bf16)f1.w;
  r[8]=(__bf16)f2.x; r[9]=(__bf16)f2.y; r[10]=(__bf16)f2.z; r[11]=(__bf16)f2.w;
  r[12]=(__bf16)f3.x;r[13]=(__bf16)f3.y;r[14]=(__bf16)f3.z; r[15]=(__bf16)f3.w;
  return r;
}

__device__ __forceinline__ v16bf load_bf16_frag(const __bf16* __restrict__ q) {
  v8bf lo = *(const v8bf*)(q);
  v8bf hi = *(const v8bf*)(q + 16);
  v16bf r;
#pragma unroll
  for (int i = 0; i < 8; ++i) { r[i] = lo[i]; r[8 + i] = hi[i]; }
  return r;
}

// ================= prep kernels =================
// h_proj[sel][b,i] = b_attn[i] + sum_j h0[b,j] * W_attn[i, j]     (columns [0,H))
__global__ __launch_bounds__(256) void k_hproj(
    const float* __restrict__ h0,
    const float* __restrict__ cW, const float* __restrict__ cb,
    const float* __restrict__ aW, const float* __restrict__ ab,
    float* __restrict__ outc, float* __restrict__ outa) {
  int tid = blockIdx.x * 256 + threadIdx.x;          // 2*B*H
  int sel = tid >= Bb * Hh;
  int rem = tid - sel * Bb * Hh;
  int i = rem >> 7, b = rem & 127;
  const float* W  = sel ? aW : cW;
  const float* bi = sel ? ab : cb;
  float* out      = sel ? outa : outc;
  const float* wrow = W + (size_t)i * (3 * Hh);
  const float* hb   = h0 + (size_t)b * Hh;
  float s = bi[i];
  for (int j = 0; j < Hh; ++j) s = fmaf(hb[j], wrow[j], s);
  out[b * Hh + i] = s;
}

// w3c[sel][i] = sum_j Wc[j] * W_attn[i, 2H + j]
__global__ __launch_bounds__(256) void k_w3c(
    const float* __restrict__ cW, const float* __restrict__ cWc,
    const float* __restrict__ aW, const float* __restrict__ aWc,
    float* __restrict__ outc, float* __restrict__ outa) {
  int tid = blockIdx.x * 256 + threadIdx.x;          // 2*H
  int sel = tid >> 9;
  int i = tid & 511;
  const float* W  = sel ? aW : cW;
  const float* Wc = sel ? aWc : cWc;
  float* out      = sel ? outa : outc;
  const float* wrow = W + (size_t)i * (3 * Hh) + 2 * Hh;
  float s = 0.f;
  for (int j = 0; j < Hh; ++j) s = fmaf(Wc[j], wrow[j], s);
  out[i] = s;
}

// w2bf[sel][n,k] = (bf16) W_attn[n, H + k]
__global__ __launch_bounds__(256) void k_w2bf(
    const float* __restrict__ cW, const float* __restrict__ aW,
    __bf16* __restrict__ outc, __bf16* __restrict__ outa) {
  int tid = blockIdx.x * 256 + threadIdx.x;          // 2*H*H
  int sel = tid >> 18;
  int rem = tid & (Hh * Hh - 1);
  int n = rem >> 9, k = rem & 511;
  const float* W = sel ? aW : cW;
  __bf16* out    = sel ? outa : outc;
  out[rem] = (__bf16)W[(size_t)n * (3 * Hh) + Hh + k];
}

__global__ __launch_bounds__(256) void k_zero(float* __restrict__ p, int n) {
  int tid = blockIdx.x * 256 + threadIdx.x;
  if (tid < n) p[tid] = 0.f;
}

// ================= attention GEMM (WMMA bf16) =================
// enc_proj[r,i] = sum_k enc[r,k]*W2[i,k];  r = t*B + b,  M = T*B = 65536, N = K = 512.
// Block = 8 waves, one 32-row M-tile; wave w owns N columns [64w, 64w+64).
// Fused epilogue: scores[b,t] += sum_i relu(acc + h_proj[b,i] + cov[b,t]*w3c[i]) * v[i]
__global__ __launch_bounds__(256) void k_attn_gemm(
    const float* __restrict__ enc,      // [T*B, H]
    const __bf16* __restrict__ w2bf,    // [H, H]
    const float* __restrict__ hproj,    // [B, H]
    const float* __restrict__ w3c,      // [H]
    const float* __restrict__ vvec,     // [H]
    const float* __restrict__ cov,      // [B, T]
    float* __restrict__ scores) {       // [B, T] pre-zeroed
  const int lane  = threadIdx.x & 31;
  const int nbase = (threadIdx.x >> 5) * 64;
  const int mbase = blockIdx.x * 32;
  const int m = lane & 15, half = lane >> 4;

  v8f acc[2][4] = {};
  for (int ks = 0; ks < 16; ++ks) {
    const int k0 = ks * 32 + 8 * half;
    v16bf a[2];
#pragma unroll
    for (int mt = 0; mt < 2; ++mt)
      a[mt] = pack_f32_frag(enc + (size_t)(mbase + 16 * mt + m) * Hh + k0);
#pragma unroll
    for (int nt = 0; nt < 4; ++nt) {
      v16bf bfrag = load_bf16_frag(w2bf + (size_t)(nbase + 16 * nt + m) * Hh + k0);
#pragma unroll
      for (int mt = 0; mt < 2; ++mt)
        acc[mt][nt] = __builtin_amdgcn_wmma_f32_16x16x32_bf16(
            false, a[mt], false, bfrag, (short)0, acc[mt][nt], false, false);
    }
  }

  // fused epilogue
  const int nl = lane & 15;
  float w3v[4], vv[4];
#pragma unroll
  for (int nt = 0; nt < 4; ++nt) {
    int i = nbase + 16 * nt + nl;
    w3v[nt] = w3c[i];
    vv[nt]  = vvec[i];
  }
#pragma unroll
  for (int mt = 0; mt < 2; ++mt) {
#pragma unroll
    for (int g = 0; g < 8; ++g) {
      int r = mbase + 16 * mt + g + 8 * half;
      int b = r & 127, t = r >> 7;
      float cvv = cov[b * Tt + t];
      float s = 0.f;
#pragma unroll
      for (int nt = 0; nt < 4; ++nt) {
        int i = nbase + 16 * nt + nl;
        float x = acc[mt][nt][g] + hproj[b * Hh + i] + cvv * w3v[nt];
        s += fmaxf(x, 0.f) * vv[nt];
      }
      s += __shfl_xor(s, 1, 32);
      s += __shfl_xor(s, 2, 32);
      s += __shfl_xor(s, 4, 32);
      s += __shfl_xor(s, 8, 32);
      if (nl == 0) atomicAdd(&scores[b * Tt + t], s);
    }
  }
}

// ================= softmax + coverage + context =================
__global__ __launch_bounds__(256) void k_softmax_ctx(
    const float* __restrict__ scores, const float* __restrict__ cov,
    const float* __restrict__ enc,
    float* __restrict__ w_out, float* __restrict__ cov_out,
    float* __restrict__ ctx) {
  __shared__ float wsh[Tt];
  __shared__ float red[256];
  int b = blockIdx.x, tid = threadIdx.x;
  float s0 = scores[b * Tt + tid], s1 = scores[b * Tt + 256 + tid];
  red[tid] = fmaxf(s0, s1);
  __syncthreads();
  for (int off = 128; off > 0; off >>= 1) {
    if (tid < off) red[tid] = fmaxf(red[tid], red[tid + off]);
    __syncthreads();
  }
  float mx = red[0];
  __syncthreads();
  float e0 = expf(s0 - mx), e1 = expf(s1 - mx);
  red[tid] = e0 + e1;
  __syncthreads();
  for (int off = 128; off > 0; off >>= 1) {
    if (tid < off) red[tid] += red[tid + off];
    __syncthreads();
  }
  float inv = 1.0f / red[0];
  float w0 = e0 * inv, w1 = e1 * inv;
  wsh[tid] = w0; wsh[256 + tid] = w1;
  w_out[b * Tt + tid] = w0;
  w_out[b * Tt + 256 + tid] = w1;
  cov_out[b * Tt + tid] = cov[b * Tt + tid] + w0;
  cov_out[b * Tt + 256 + tid] = cov[b * Tt + 256 + tid] + w1;
  __syncthreads();
  for (int h = tid; h < Hh; h += 256) {
    float acc = 0.f;
    for (int t = 0; t < Tt; ++t)
      acc = fmaf(wsh[t], enc[((size_t)t * Bb + b) * Hh + h], acc);
    ctx[b * Hh + h] = acc;
  }
}

__global__ __launch_bounds__(256) void k_ctx_comb(
    const float* __restrict__ cc, const float* __restrict__ ca,
    float* __restrict__ out) {
  int tid = blockIdx.x * 256 + threadIdx.x;          // B*H
  out[tid] = 0.5f * (cc[tid] + ca[tid]);
}

// ================= GRU step =================
__global__ __launch_bounds__(256) void k_gru(
    const int* __restrict__ inputs, const float* __restrict__ emb_table,
    const float* __restrict__ ctx, const float* __restrict__ h0,
    const float* __restrict__ W_ih, const float* __restrict__ W_hh,
    const float* __restrict__ b_ih, const float* __restrict__ b_hh,
    float* __restrict__ h1_out) {
  int tid = blockIdx.x * 256 + threadIdx.x;          // B*H
  int h = tid >> 7, b = tid & 127;                   // wave-uniform h
  const float* e  = emb_table + (size_t)inputs[b] * Ee;
  const float* cb = ctx + (size_t)b * Hh;
  const float* hb = h0 + (size_t)b * Hh;
  const int KI = Ee + Hh;
  const float* wr = W_ih + (size_t)h * KI;
  const float* wz = W_ih + (size_t)(Hh + h) * KI;
  const float* wn = W_ih + (size_t)(2 * Hh + h) * KI;
  float ir = b_ih[h], iz = b_ih[Hh + h], inn = b_ih[2 * Hh + h];
  for (int k = 0; k < Ee; ++k) {
    float x = e[k];
    ir = fmaf(x, wr[k], ir); iz = fmaf(x, wz[k], iz); inn = fmaf(x, wn[k], inn);
  }
  for (int k = 0; k < Hh; ++k) {
    float x = cb[k];
    ir = fmaf(x, wr[Ee + k], ir); iz = fmaf(x, wz[Ee + k], iz); inn = fmaf(x, wn[Ee + k], inn);
  }
  const float* ur = W_hh + (size_t)h * Hh;
  const float* uz = W_hh + (size_t)(Hh + h) * Hh;
  const float* un = W_hh + (size_t)(2 * Hh + h) * Hh;
  float hr = b_hh[h], hz = b_hh[Hh + h], hn = b_hh[2 * Hh + h];
  for (int k = 0; k < Hh; ++k) {
    float y = hb[k];
    hr = fmaf(y, ur[k], hr); hz = fmaf(y, uz[k], hz); hn = fmaf(y, un[k], hn);
  }
  float r = 1.f / (1.f + expf(-(ir + hr)));
  float z = 1.f / (1.f + expf(-(iz + hz)));
  float n = tanhf(inn + r * hn);
  h1_out[b * Hh + h] = (1.f - z) * n + z * hb[h];
}

// xcat_bf[b, k] = bf16( k<H ? h1[b,k] : ctx[b,k-H] )
__global__ __launch_bounds__(256) void k_xcat(
    const float* __restrict__ h1, const float* __restrict__ ctx,
    __bf16* __restrict__ xbf) {
  int tid = blockIdx.x * 256 + threadIdx.x;          // B*2H
  int b = tid >> 10, k = tid & 1023;
  float v = (k < Hh) ? h1[b * Hh + k] : ctx[b * Hh + (k - Hh)];
  xbf[tid] = (__bf16)v;
}

// ================= output GEMM (WMMA bf16): logits = [h1,ctx] @ W_out^T + b_out ====
// M = 128 (full per wave), N-tile = 16 per wave, K = 1024.  1875 waves exactly.
__global__ __launch_bounds__(128) void k_out_gemm(
    const __bf16* __restrict__ xbf,   // [128, 1024]
    const float* __restrict__ Wout,   // [V_OUT, 1024]
    const float* __restrict__ bout,   // [V_OUT]
    float* __restrict__ logits) {     // [128, V_OUT]
  int wid = blockIdx.x * 4 + (threadIdx.x >> 5);
  if (wid >= Vout / 16) return;       // wave-uniform exit
  int lane = threadIdx.x & 31;
  int nbase = wid * 16;
  int m = lane & 15, half = lane >> 4;
  v8f acc[8] = {};
  for (int ks = 0; ks < 32; ++ks) {
    int k0 = ks * 32 + 8 * half;
    v16bf bfrag = pack_f32_frag(Wout + (size_t)(nbase + m) * 1024 + k0);
#pragma unroll
    for (int mt = 0; mt < 8; ++mt) {
      v16bf afrag = load_bf16_frag(xbf + (size_t)(16 * mt + m) * 1024 + k0);
      acc[mt] = __builtin_amdgcn_wmma_f32_16x16x32_bf16(
          false, afrag, false, bfrag, (short)0, acc[mt], false, false);
    }
  }
  int nl = lane & 15;
  int o = nbase + nl;
  float bo = bout[o];
#pragma unroll
  for (int mt = 0; mt < 8; ++mt) {
#pragma unroll
    for (int g = 0; g < 8; ++g) {
      int row = 16 * mt + g + 8 * half;
      logits[(size_t)row * Vout + o] = acc[mt][g] + bo;
    }
  }
}

// in-place log_softmax over V_OUT per row
__global__ __launch_bounds__(256) void k_logsoftmax(float* __restrict__ lg) {
  __shared__ float red[256];
  int b = blockIdx.x, tid = threadIdx.x;
  float* row = lg + (size_t)b * Vout;
  float mx = -3.402823466e38f;
  for (int o = tid; o < Vout; o += 256) mx = fmaxf(mx, row[o]);
  red[tid] = mx;
  __syncthreads();
  for (int off = 128; off > 0; off >>= 1) {
    if (tid < off) red[tid] = fmaxf(red[tid], red[tid + off]);
    __syncthreads();
  }
  mx = red[0];
  __syncthreads();
  float sum = 0.f;
  for (int o = tid; o < Vout; o += 256) sum += expf(row[o] - mx);
  red[tid] = sum;
  __syncthreads();
  for (int off = 128; off > 0; off >>= 1) {
    if (tid < off) red[tid] += red[tid + off];
    __syncthreads();
  }
  float lse = mx + logf(red[0]);
  __syncthreads();
  for (int o = tid; o < Vout; o += 256) row[o] = row[o] - lse;
}

// ================= launch =================
extern "C" void kernel_launch(void* const* d_in, const int* in_sizes, int n_in,
                              void* d_out, int out_size, void* d_ws, size_t ws_size,
                              hipStream_t stream) {
  (void)in_sizes; (void)n_in; (void)out_size; (void)ws_size;
  const int*   inputs    = (const int*)  d_in[0];
  const float* last_h    = (const float*)d_in[1];
  const float* code_out  = (const float*)d_in[2];
  const float* ast_out   = (const float*)d_in[3];
  const float* code_cov  = (const float*)d_in[4];
  const float* ast_cov   = (const float*)d_in[5];
  const float* embedding = (const float*)d_in[6];
  const float* cW        = (const float*)d_in[7];
  const float* cb        = (const float*)d_in[8];
  const float* cv        = (const float*)d_in[9];
  const float* cWc       = (const float*)d_in[10];
  const float* aW        = (const float*)d_in[11];
  const float* ab        = (const float*)d_in[12];
  const float* av        = (const float*)d_in[13];
  const float* aWc       = (const float*)d_in[14];
  const float* gWih      = (const float*)d_in[15];
  const float* gWhh      = (const float*)d_in[16];
  const float* gbih      = (const float*)d_in[17];
  const float* gbhh      = (const float*)d_in[18];
  const float* Wout      = (const float*)d_in[19];
  const float* bout      = (const float*)d_in[20];

  char* ws = (char*)d_ws;
  __bf16* w2c   = (__bf16*)(ws + WS_W2C);
  __bf16* w2a   = (__bf16*)(ws + WS_W2A);
  float*  hpc   = (float*)(ws + WS_HPC);
  float*  hpa   = (float*)(ws + WS_HPA);
  float*  w3c_c = (float*)(ws + WS_W3C);
  float*  w3c_a = (float*)(ws + WS_W3A);
  float*  scc   = (float*)(ws + WS_SCC);
  float*  sca   = (float*)(ws + WS_SCA);
  float*  ctxc  = (float*)(ws + WS_CTXC);
  float*  ctxa  = (float*)(ws + WS_CTXA);
  float*  ctx   = (float*)(ws + WS_CTX);
  __bf16* xbf   = (__bf16*)(ws + WS_XBF);

  float* out     = (float*)d_out;
  float* o_logsm = out;                       // [B, V_OUT]
  float* o_h1    = o_logsm + (size_t)Bb * Vout;
  float* o_cw    = o_h1 + Bb * Hh;
  float* o_aw    = o_cw + Bb * Tt;
  float* o_ccov  = o_aw + Bb * Tt;
  float* o_acov  = o_ccov + Bb * Tt;

  // prep
  k_hproj<<<(2 * Bb * Hh) / 256, 256, 0, stream>>>(last_h, cW, cb, aW, ab, hpc, hpa);
  k_w3c<<<(2 * Hh) / 256, 256, 0, stream>>>(cW, cWc, aW, aWc, w3c_c, w3c_a);
  k_w2bf<<<(2 * Hh * Hh) / 256, 256, 0, stream>>>(cW, aW, w2c, w2a);
  k_zero<<<(2 * Bb * Tt) / 256, 256, 0, stream>>>(scc, 2 * Bb * Tt);  // scc+sca contiguous

  // attention energy GEMMs with fused relu-dot-v epilogue -> scores
  k_attn_gemm<<<(Tt * Bb) / 32, 256, 0, stream>>>(code_out, w2c, hpc, w3c_c, cv, code_cov, scc);
  k_attn_gemm<<<(Tt * Bb) / 32, 256, 0, stream>>>(ast_out,  w2a, hpa, w3c_a, av, ast_cov,  sca);

  // softmax / coverage / context
  k_softmax_ctx<<<Bb, 256, 0, stream>>>(scc, code_cov, code_out, o_cw, o_ccov, ctxc);
  k_softmax_ctx<<<Bb, 256, 0, stream>>>(sca, ast_cov,  ast_out,  o_aw, o_acov, ctxa);
  k_ctx_comb<<<(Bb * Hh) / 256, 256, 0, stream>>>(ctxc, ctxa, ctx);

  // GRU -> h1 (written straight to d_out)
  k_gru<<<(Bb * Hh) / 256, 256, 0, stream>>>(inputs, embedding, ctx, last_h,
                                             gWih, gWhh, gbih, gbhh, o_h1);

  // output projection + log_softmax (in-place on d_out)
  k_xcat<<<(Bb * 2 * Hh) / 256, 256, 0, stream>>>(o_h1, ctx, xbf);
  k_out_gemm<<<(Vout / 16 + 3) / 4, 128, 0, stream>>>(xbf, Wout, bout, o_logsm);
  k_logsoftmax<<<Bb, 256, 0, stream>>>(o_logsm);
}
